// PolicyDecoderBase_69715909149436
// MI455X (gfx1250) — matrix operand order repaired and checked by reading, not compiled, not run on hardware
//
#include <hip/hip_runtime.h>
#include <math.h>

#define BB 64
#define NN 2048
#define DD 512
#define HH 16
#define KSZ 32
#define HK 512   // H*KS

typedef __bf16  bf16x16 __attribute__((ext_vector_type(16)));
typedef __bf16  bf16x8  __attribute__((ext_vector_type(8)));
typedef float   f32x8   __attribute__((ext_vector_type(8)));
typedef unsigned int u32x4 __attribute__((ext_vector_type(4)));
typedef int     i32x8   __attribute__((ext_vector_type(8)));
typedef int     i32x4   __attribute__((ext_vector_type(4)));

#if __has_builtin(__builtin_amdgcn_tensor_load_to_lds)
#define HAVE_TDM 1
#endif

// LDS staging geometry for phase 2 (padded rows: 512 + 4 pad dwords / row)
// pad: 2 dwords every 256 dwords -> row stride 516 dwords = 2064 B (16B aligned,
// bank stride 4 -> ds_load_b128 spreads across banks instead of 32-way conflict)
#define STAGE_ROW_DW 516
#define STAGE_BYTES  66048              // 32*512*4 + (32*512/256)*2*4
#define PROBS_OFF    (2 * STAGE_BYTES)  // 132096
#define SCRATCH_OFF  (PROBS_OFF + NN * HH * 2)
#define SMEM_BYTES   (SCRATCH_OFF + 1280)

// ---------------------------------------------------------------------------
// WMMA fragment helpers (CDNA5 §7.12.2 wave32 layouts)
// ---------------------------------------------------------------------------

// A 16x32 bf16 from row-major f32. Lane L: m=L%16, half=L/16; the K-pair
// pattern collapses to two contiguous 8-float runs: [half*8, +8) and [16+half*8, +8)
__device__ __forceinline__ bf16x16 load_a_f32(const float* __restrict__ src, int ld) {
  const int lane = threadIdx.x & 31;
  const int m    = lane & 15;
  const int half = lane >> 4;
  const float* row = src + (size_t)m * ld + half * 8;
  const float4 x0 = *(const float4*)(row);
  const float4 x1 = *(const float4*)(row + 4);
  const float4 x2 = *(const float4*)(row + 16);
  const float4 x3 = *(const float4*)(row + 20);
  bf16x16 f;
  f[0]=(__bf16)x0.x;  f[1]=(__bf16)x0.y;  f[2]=(__bf16)x0.z;  f[3]=(__bf16)x0.w;
  f[4]=(__bf16)x1.x;  f[5]=(__bf16)x1.y;  f[6]=(__bf16)x1.z;  f[7]=(__bf16)x1.w;
  f[8]=(__bf16)x2.x;  f[9]=(__bf16)x2.y;  f[10]=(__bf16)x2.z; f[11]=(__bf16)x2.w;
  f[12]=(__bf16)x3.x; f[13]=(__bf16)x3.y; f[14]=(__bf16)x3.z; f[15]=(__bf16)x3.w;
  return f;
}

// A 16x32 bf16 from probs stored [h][n] (row length NN): two ds_load_b128s.
__device__ __forceinline__ bf16x16 load_a_probsT(const __bf16* __restrict__ p, int kbase) {
  const int lane = threadIdx.x & 31;
  const int m    = lane & 15;
  const int half = lane >> 4;
  const __bf16* row = p + (size_t)m * NN + kbase + half * 8;
  const bf16x8 lo = *(const bf16x8*)(row);
  const bf16x8 hi = *(const bf16x8*)(row + 16);
  bf16x16 f;
#pragma unroll
  for (int i = 0; i < 8; ++i) { f[i] = lo[i]; f[8 + i] = hi[i]; }
  return f;
}

// B 32x16 bf16 from row-major f32 (lane L: n=L%16, K runs [half*8,+8) and [16+half*8,+8)).
__device__ __forceinline__ bf16x16 load_b_f32(const float* __restrict__ src, int ld) {
  const int lane = threadIdx.x & 31;
  const int n    = lane & 15;
  const int half = lane >> 4;
  bf16x16 f;
#pragma unroll
  for (int i = 0; i < 8; ++i) {
    f[i]     = (__bf16)src[(size_t)(half * 8 + i) * ld + n];
    f[8 + i] = (__bf16)src[(size_t)(16 + half * 8 + i) * ld + n];
  }
  return f;
}

// B 32x16 bf16 from the padded LDS staging buffer.
__device__ __forceinline__ bf16x16 load_b_stage(const float* __restrict__ st, int col0) {
  const int lane = threadIdx.x & 31;
  const int half = lane >> 4;
  const float* col = st + col0 + (col0 >> 8) * 2 + (lane & 15);
  bf16x16 f;
#pragma unroll
  for (int i = 0; i < 8; ++i) {
    f[i]     = (__bf16)col[(size_t)(half * 8 + i) * STAGE_ROW_DW];
    f[8 + i] = (__bf16)col[(size_t)(16 + half * 8 + i) * STAGE_ROW_DW];
  }
  return f;
}

// C/D 16x16 f32: lane L -> col n = L%16, rows m = (L/16)*8 + r.
__device__ __forceinline__ void store_d(float* __restrict__ dst, int ld, f32x8 acc) {
  const int lane  = threadIdx.x & 31;
  const int n     = lane & 15;
  const int mbase = (lane >> 4) * 8;
#pragma unroll
  for (int r = 0; r < 8; ++r) dst[(size_t)(mbase + r) * ld + n] = acc[r];
}

__device__ __forceinline__ f32x8 wmma_bf16(bf16x16 a, bf16x16 b, f32x8 c) {
  return __builtin_amdgcn_wmma_f32_16x16x32_bf16(false, a, false, b, (short)0, c,
                                                 false, false);
}

#ifdef HAVE_TDM
// Issue one TDM descriptor: DMA a 32x512 f32 tile (row stride 512) from global
// into LDS at lds_off, with 2-dword padding every 256 dwords (ISA §8.3-8.4).
// This toolchain's builtin takes 6 args: (g0, g1, g2, g3, g4, cpol).
__device__ __forceinline__ void tdm_load_tile(const float* gsrc, unsigned lds_off) {
  const unsigned long long ga = (unsigned long long)gsrc;
  u32x4 g0;
  g0[0] = 1u;                                            // count=1, user desc
  g0[1] = lds_off;                                       // lds_addr (bytes)
  g0[2] = (unsigned)(ga & 0xffffffffu);                  // global_addr[31:0]
  g0[3] = (unsigned)((ga >> 32) & 0x1ffffffu) | (2u << 30); // addr[56:32], type=2
  i32x8 g1;
  g1[0] = 0x20000            // data_size = 2 (4 bytes)
        | (1 << 20)          // pad_enable
        | (7 << 22)          // pad_interval = 256 dwords
        | (1 << 25);         // pad_amount  = 2 dwords
  g1[1] = (int)(512u << 16); // tensor_dim0 = 512 (bits 79:48, low half)
  g1[2] = (int)(2048u << 16);// tensor_dim1 = 2048
  g1[3] = (int)(512u << 16); // tile_dim0 = 512 (bits 127:112)
  g1[4] = 32;                // tile_dim1 = 32
  g1[5] = 512;               // tensor_dim0_stride = 512
  g1[6] = 0;
  g1[7] = 0;
  const i32x4 z4 = {0, 0, 0, 0};
  const i32x8 z8 = {0, 0, 0, 0, 0, 0, 0, 0};
  __builtin_amdgcn_tensor_load_to_lds(g0, g1, z4, z4, z8, 0);
}
__device__ __forceinline__ void tdm_wait0() {
#if __has_builtin(__builtin_amdgcn_s_wait_tensorcnt)
  __builtin_amdgcn_s_wait_tensorcnt(0);
#else
  asm volatile("s_wait_tensorcnt 0x0" ::: "memory");
#endif
}
#endif

// ---------------------------------------------------------------------------
// Kernel 1: q[b,:] = context_b @ wq + bq  (64x512, K=512) via WMMA
// ---------------------------------------------------------------------------
__global__ void qproj_kernel(const float* __restrict__ ctxin,
                             const float* __restrict__ wq,
                             const float* __restrict__ bq,
                             float* __restrict__ qout) {
  const int nt = blockIdx.x;
  const int mt = blockIdx.y;
  f32x8 acc = {};
  for (int kk = 0; kk < DD; kk += 32) {
    bf16x16 a = load_a_f32(ctxin + (size_t)(mt * 16) * DD + kk, DD);
    bf16x16 b = load_b_f32(wq + (size_t)kk * HK + nt * 16, HK);
    acc = wmma_bf16(a, b, acc);
  }
  const float bias = bq[nt * 16 + (threadIdx.x & 15)];
#pragma unroll
  for (int r = 0; r < 8; ++r) acc[r] += bias;
  store_d(qout + (size_t)(mt * 16) * HK + nt * 16, HK, acc);
}

// ---------------------------------------------------------------------------
// Kernel 2: fold q into wk (+ bias term of logits)
// ---------------------------------------------------------------------------
__global__ void fold_kernel(const float* __restrict__ wk,
                            const float* __restrict__ bk,
                            const float* __restrict__ qv,
                            float* __restrict__ wkq,
                            float* __restrict__ qbk) {
  const int idx = blockIdx.x * 256 + threadIdx.x;   // over B*D*H
  const int b   = idx / (DD * HH);
  const int rem = idx % (DD * HH);
  const int d   = rem / HH;
  const int h   = rem % HH;
  const float* qh   = qv + (size_t)b * HK + h * KSZ;
  const float* wrow = wk + (size_t)d * HK + h * KSZ;
  float s = 0.f;
#pragma unroll
  for (int c = 0; c < KSZ; ++c) s = fmaf(wrow[c], qh[c], s);
  wkq[(size_t)b * DD * HH + (size_t)d * HH + h] = s;
  if (d == 0) {
    float t = 0.f;
#pragma unroll
    for (int c = 0; c < KSZ; ++c) t = fmaf(bk[h * KSZ + c], qh[c], t);
    qbk[b * HH + h] = t;
  }
}

// ---------------------------------------------------------------------------
// Kernel 3: fused per-batch attention (one WG / batch, 8 waves, ~199 KB LDS)
// ---------------------------------------------------------------------------
__global__ void attn_kernel(const float* __restrict__ emb,
                            const unsigned char* __restrict__ mask,
                            const float* __restrict__ wkq,
                            const float* __restrict__ qbk,
                            float* __restrict__ u,
                            float* __restrict__ lsum) {
  const int b    = blockIdx.x;
  const int wave = threadIdx.x >> 5;
  const int lane = threadIdx.x & 31;

  extern __shared__ char smem[];
  float*  logits  = (float*)smem;                       // N*H f32 (phase 1 only)
  __bf16* probsT  = (__bf16*)(smem + PROBS_OFF);        // [H][N] bf16
  float*  scratch = (float*)(smem + SCRATCH_OFF);

  const float* embb = emb + (size_t)b * NN * DD;
  const float* wkqb = wkq + (size_t)b * DD * HH;
  const float  rscale = 0.17677669529663687f;           // 1/sqrt(32)

  // ---- phase 1: masked logits.  wkq B-panel register-resident (16 frags) ----
  bf16x16 bfr[16];
#pragma unroll
  for (int kk = 0; kk < 16; ++kk)
    bfr[kk] = load_b_f32(wkqb + (size_t)(kk * 32) * HH, HH);

  for (int mt = 0; mt < 16; ++mt) {
    const int row0 = wave * 256 + mt * 16;
    f32x8 acc = {};
#pragma unroll
    for (int kk = 0; kk < 16; ++kk) {
      __builtin_prefetch(embb + (size_t)(row0 + 16) * DD + kk * 32, 0, 0);
      bf16x16 a = load_a_f32(embb + (size_t)row0 * DD + kk * 32, DD);
      acc = wmma_bf16(a, bfr[kk], acc);
    }
    const int h     = lane & 15;
    const int mbase = (lane >> 4) * 8;
    const float cb  = qbk[b * HH + h];
#pragma unroll
    for (int r = 0; r < 8; ++r) {
      const int row = row0 + mbase + r;
      float lg = (acc[r] + cb) * rscale;
      if (!mask[(size_t)b * NN + row]) lg = -1e30f;
      logits[row * HH + h] = lg;
    }
  }
  __syncthreads();

  // ---- softmax over N per head (16 threads per head; head = tid>>4) ----
  const int hh  = threadIdx.x >> 4;
  const int sub = threadIdx.x & 15;
  float mx = -3.4e38f;
  for (int n = sub; n < NN; n += 16) mx = fmaxf(mx, logits[n * HH + hh]);
  scratch[hh * 16 + sub] = mx;
  __syncthreads();
  if (sub == 0) {
    float mm = -3.4e38f;
#pragma unroll
    for (int i = 0; i < 16; ++i) mm = fmaxf(mm, scratch[hh * 16 + i]);
    scratch[256 + hh] = mm;
  }
  __syncthreads();
  const float hm = scratch[256 + hh];
  float ssum = 0.f;
  for (int n = sub; n < NN; n += 16) {
    const float p = __expf(logits[n * HH + hh] - hm);
    probsT[(size_t)hh * NN + n] = (__bf16)p;
    ssum += p;
  }
  scratch[hh * 16 + sub] = ssum;
  __syncthreads();
  if (sub == 0) {
    float s = 0.f;
#pragma unroll
    for (int i = 0; i < 16; ++i) s += scratch[hh * 16 + i];
    lsum[b * HH + hh] = s;
  }
  __syncthreads();   // logits region now dead -> reused as TDM staging

  // ---- phase 2: u = probsT(16x2048) @ emb(2048x512), wave owns 64 cols ----
  f32x8 acc2[4] = {};
#ifdef HAVE_TDM
  const unsigned st_off0 = (unsigned)(unsigned long long)(void*)smem;
  if (wave == 0) tdm_load_tile(embb, st_off0);
  for (int k0 = 0; k0 < NN; k0 += 32) {
    const int bufi = (k0 >> 5) & 1;
    if (wave == 0) tdm_wait0();             // current tile landed (TENSORcnt)
    __syncthreads();                        // ...and visible to all 8 waves
    if (wave == 0 && k0 + 32 < NN)          // kick DMA of the next tile
      tdm_load_tile(embb + (size_t)(k0 + 32) * DD,
                    st_off0 + (unsigned)((bufi ^ 1) * STAGE_BYTES));
    const float* st = (const float*)(smem + (size_t)bufi * STAGE_BYTES);
    bf16x16 a = load_a_probsT(probsT, k0);
#pragma unroll
    for (int nt = 0; nt < 4; ++nt) {
      const int col0 = (wave * 4 + nt) * 16;
      bf16x16 bb = load_b_stage(st, col0);
      acc2[nt] = wmma_bf16(a, bb, acc2[nt]);
    }
  }
#else
  for (int k0 = 0; k0 < NN; k0 += 32) {
    bf16x16 a = load_a_probsT(probsT, k0);
#pragma unroll
    for (int nt = 0; nt < 4; ++nt) {
      const int col0 = (wave * 4 + nt) * 16;
      bf16x16 bb = load_b_f32(embb + (size_t)k0 * DD + col0, DD);
      acc2[nt] = wmma_bf16(a, bb, acc2[nt]);
    }
  }
#endif
  float* ub = u + (size_t)b * HH * DD;
#pragma unroll
  for (int nt = 0; nt < 4; ++nt) {
    const int col0 = (wave * 4 + nt) * 16;
    store_d(ub + col0, DD, acc2[nt]);       // element (h, e) -> u[b][h*D + e]
  }
}

// ---------------------------------------------------------------------------
// Kernel 4: per-batch output projection (tiny)
// ---------------------------------------------------------------------------
__global__ void outproj_kernel(const float* __restrict__ u,
                               const float* __restrict__ lsum,
                               const float* __restrict__ wv,
                               const float* __restrict__ bv,
                               const float* __restrict__ wo,
                               const float* __restrict__ bo,
                               float* __restrict__ ctxw) {
  const int b = blockIdx.x;
  __shared__ float o[HK];
  const float* ub = u + (size_t)b * HH * DD;
  for (int j = threadIdx.x; j < HK; j += 256) {
    const int h = j / KSZ;
    const int d = j % KSZ;
    const float inv = 1.f / lsum[b * HH + h];
    float s = 0.f;
    for (int e = 0; e < DD; ++e)
      s = fmaf(ub[(size_t)h * DD + e], wv[(size_t)e * HK + h * KSZ + d], s);
    o[j] = s * inv + bv[j];                 // sum(attn)==1 after normalization
  }
  __syncthreads();
  for (int e = threadIdx.x; e < DD; e += 256) {
    float s = bo[e];
    for (int j = 0; j < HK; ++j) s = fmaf(o[j], wo[(size_t)j * DD + e], s);
    ctxw[(size_t)b * DD + e] = s;
  }
}

// ---------------------------------------------------------------------------
// Kernel 5: scoring pass (second full read of embeddings), memory bound
// ---------------------------------------------------------------------------
__global__ void score_kernel(const float* __restrict__ emb,
                             const float* __restrict__ ctxw,
                             const unsigned char* __restrict__ mask,
                             float* __restrict__ out) {
  const int gid  = blockIdx.x * 8 + (threadIdx.x >> 5);  // row in [0, B*N)
  const int lane = threadIdx.x & 31;
  const int b    = gid / NN;
  const float4* e = (const float4*)(emb + (size_t)gid * DD);
  const float4* c = (const float4*)(ctxw + (size_t)b * DD);
  float s = 0.f;
#pragma unroll
  for (int i = lane; i < DD / 4; i += 32) {
    const float4 x = e[i];
    const float4 y = c[i];
    s = fmaf(x.x, y.x, s);
    s = fmaf(x.y, y.y, s);
    s = fmaf(x.z, y.z, s);
    s = fmaf(x.w, y.w, s);
  }
#pragma unroll
  for (int off = 16; off > 0; off >>= 1) s += __shfl_xor(s, off, 32);
  if (lane == 0) {
    const float al = 10.f * tanhf(s * 0.04419417382415922f);  // 1/sqrt(512)
    out[gid] = mask[gid] ? al : -3.4028234663852886e38f;
  }
}

// ---------------------------------------------------------------------------
extern "C" void kernel_launch(void* const* d_in, const int* in_sizes, int n_in,
                              void* d_out, int out_size, void* d_ws, size_t ws_size,
                              hipStream_t stream) {
  const float* ctxin = (const float*)d_in[0];
  const float* emb   = (const float*)d_in[1];
  const unsigned char* mask = (const unsigned char*)d_in[2];
  const float* wq = (const float*)d_in[3];
  const float* bq = (const float*)d_in[4];
  const float* wk = (const float*)d_in[5];
  const float* bk = (const float*)d_in[6];
  const float* wv = (const float*)d_in[7];
  const float* bv = (const float*)d_in[8];
  const float* wo = (const float*)d_in[9];
  const float* bo = (const float*)d_in[10];
  float* out = (float*)d_out;

  float* ws   = (float*)d_ws;
  float* qw   = ws;                              // B*HK
  float* wkq  = qw  + (size_t)BB * HK;           // B*D*H
  float* qbk  = wkq + (size_t)BB * DD * HH;      // B*H
  float* u    = qbk + (size_t)BB * HH;           // B*H*D
  float* lsum = u   + (size_t)BB * HH * DD;      // B*H
  float* ctxw = lsum + (size_t)BB * HH;          // B*D

  qproj_kernel<<<dim3(HK / 16, BB / 16), 32, 0, stream>>>(ctxin, wq, bq, qw);
  fold_kernel<<<(BB * DD * HH) / 256, 256, 0, stream>>>(wk, bk, qw, wkq, qbk);
  attn_kernel<<<BB, 256, SMEM_BYTES, stream>>>(emb, mask, wkq, qbk, u, lsum);
  outproj_kernel<<<BB, 256, 0, stream>>>(u, lsum, wv, bv, wo, bo, ctxw);
  score_kernel<<<(BB * NN) / 8, 256, 0, stream>>>(emb, ctxw, mask, out);
}
